// GNNBackbone_53034256171351
// MI455X (gfx1250) — matrix operand order repaired and checked by reading, not compiled, not run on hardware
//
#include <hip/hip_runtime.h>

#define HDIM 128
#define LPAD 132          // padded LDS row stride (floats) to avoid bank conflicts
#define EPSV 1e-5f

typedef __attribute__((ext_vector_type(16))) _Float16 v16h;
typedef __attribute__((ext_vector_type(8)))  float    v8f;

// ---------------------------------------------------------------------------
// Zero the scatter accumulators (agg [n*H] and sumw [n] are contiguous in ws)
// ---------------------------------------------------------------------------
__global__ void gnn_zero(float* __restrict__ p, long long cnt) {
  long long i  = (long long)blockIdx.x * blockDim.x + threadIdx.x;
  long long st = (long long)gridDim.x * blockDim.x;
  for (; i < cnt; i += st) p[i] = 0.0f;
}

// ---------------------------------------------------------------------------
// Edge scatter: one wave per edge; lane l handles features [4l, 4l+4)
//   agg[dst] += w * h[src];   sumw[dst] += w
// ---------------------------------------------------------------------------
__global__ void gnn_scatter(const float* __restrict__ h,
                            const int*  __restrict__ src,
                            const int*  __restrict__ dst,
                            const float* __restrict__ ew,
                            float* __restrict__ agg,
                            float* __restrict__ sumw,
                            int nedge) {
  const int lane = threadIdx.x & 31;
  long long wv = ((long long)blockIdx.x * blockDim.x + threadIdx.x) >> 5;
  long long nw = ((long long)gridDim.x * blockDim.x) >> 5;
  for (long long e = wv; e < nedge; e += nw) {
    const int   s = src[e];
    const int   d = dst[e];
    const float w = ew[e];
    const float4 v = ((const float4*)(h + (long long)s * HDIM))[lane];
    float* ap = agg + (long long)d * HDIM + lane * 4;
    atomicAdd(ap + 0, w * v.x);
    atomicAdd(ap + 1, w * v.y);
    atomicAdd(ap + 2, w * v.z);
    atomicAdd(ap + 3, w * v.w);
    if (lane == 0) atomicAdd(sumw + d, w);
  }
}

// ---------------------------------------------------------------------------
// Fused per-layer kernel: out = LN(relu((agg/sumw)@Wn^T + h@Wr^T + b))
// Block = 256 threads (8 waves) handles 16 rows x 128 cols.
// Wave w computes the 16x16 output tile for cols [16w,16w+16) with
// v_wmma_f32_16x16x32_f16 (4 K-chunks x 2 GEMMs, shared f32 accumulator).
// ---------------------------------------------------------------------------
__global__ void __launch_bounds__(256) gnn_fused(
    const float* __restrict__ h,    const float* __restrict__ agg,
    const float* __restrict__ sumw,
    const float* __restrict__ wn,   const float* __restrict__ wr,
    const float* __restrict__ br,   const float* __restrict__ gam,
    const float* __restrict__ bet,  float* __restrict__ out, int n) {
  __shared__ float sN[16 * LPAD];   // scaled neighbor rows (agg / max(sumw,1))
  __shared__ float sR[16 * LPAD];   // root rows (h)

  const int tid  = threadIdx.x;
  const int wid  = tid >> 5;        // wave 0..7 -> column tile
  const int lane = tid & 31;
  const int l16  = lane & 15;
  const int hk   = lane >> 4;       // half-wave select
  const int rowbase = blockIdx.x * 16;

  // --- stage 16x128 tiles into LDS (fully coalesced float4 global loads) ---
  for (int i = tid; i < 16 * (HDIM / 4); i += 256) {
    const int rr = i >> 5;          // tile row 0..15
    const int cc = (i & 31) * 4;    // col 0,4,...,124
    int grow = rowbase + rr; if (grow >= n) grow = n - 1;
    const float4 hv = *(const float4*)(h   + (long long)grow * HDIM + cc);
    float4       av = *(const float4*)(agg + (long long)grow * HDIM + cc);
    const float sc = 1.0f / fmaxf(sumw[grow], 1.0f);
    av.x *= sc; av.y *= sc; av.z *= sc; av.w *= sc;
    *(float4*)(sR + rr * LPAD + cc) = hv;
    *(float4*)(sN + rr * LPAD + cc) = av;
  }
  __syncthreads();

  const int col = wid * 16 + l16;                     // output column 0..127
  const float* wnr = wn + (long long)col * HDIM;      // row of Wn -> B column
  const float* wrr = wr + (long long)col * HDIM;
  const float* aN  = sN + l16 * LPAD;                 // A-matrix row for lane
  const float* aR  = sR + l16 * LPAD;

  v8f acc = {};
#pragma unroll
  for (int c = 0; c < 4; ++c) {                       // K chunks of 32
    const int kb = c * 32;
    v16h a, b;
    // A layout (16-bit 16x32): lanes 0-15 hold K {kb..kb+7, kb+16..kb+23},
    // lanes 16-31 hold the +8 offsets.
#pragma unroll
    for (int t = 0; t < 8; ++t) {
      a[t]     = (_Float16)aN[kb +      hk * 8 + t];
      a[t + 8] = (_Float16)aN[kb + 16 + hk * 8 + t];
    }
    // B layout (32x16): same K order as A: lanes 0-15 hold K {kb..kb+7, kb+16..kb+23},
    // lanes 16-31 the +8 offsets. Row-major W (= W^T column).
#pragma unroll
    for (int t = 0; t < 8; ++t) {
      b[t]     = (_Float16)wnr[kb +      hk * 8 + t];
      b[t + 8] = (_Float16)wnr[kb + 16 + hk * 8 + t];
    }
    acc = __builtin_amdgcn_wmma_f32_16x16x32_f16(false, a, false, b,
                                                 (short)0, acc, false, false);
#pragma unroll
    for (int t = 0; t < 8; ++t) {
      a[t]     = (_Float16)aR[kb +      hk * 8 + t];
      a[t + 8] = (_Float16)aR[kb + 16 + hk * 8 + t];
    }
#pragma unroll
    for (int t = 0; t < 8; ++t) {
      b[t]     = (_Float16)wrr[kb +      hk * 8 + t];
      b[t + 8] = (_Float16)wrr[kb + 16 + hk * 8 + t];
    }
    acc = __builtin_amdgcn_wmma_f32_16x16x32_f16(false, a, false, b,
                                                 (short)0, acc, false, false);
  }
  __syncthreads();   // all waves done reading sN/sR before reuse

  // --- bias + relu, deposit D (VGPR v -> row v + 8*hk, col = lane%16) ------
  const float bias = br[col];
#pragma unroll
  for (int v = 0; v < 8; ++v) {
    const int orow = v + hk * 8;
    sN[orow * LPAD + col] = fmaxf(acc[v] + bias, 0.0f);
  }
  __syncthreads();

  // --- per-row LayerNorm over 128 features ---------------------------------
  // wave w: lanes 0-15 -> row 2w, lanes 16-31 -> row 2w+1; 8 feats per lane
  const int lrow = 2 * wid + hk;
  const float* lp = sN + lrow * LPAD + l16 * 8;
  float vals[8];
  float s = 0.0f, ss = 0.0f;
#pragma unroll
  for (int t = 0; t < 8; ++t) {
    vals[t] = lp[t];
    s  += vals[t];
    ss += vals[t] * vals[t];
  }
#pragma unroll
  for (int off = 1; off < 16; off <<= 1) {   // reduce within each 16-lane half
    s  += __shfl_xor(s,  off, 32);
    ss += __shfl_xor(ss, off, 32);
  }
  const float mu   = s * (1.0f / HDIM);
  const float var  = fmaxf(ss * (1.0f / HDIM) - mu * mu, 0.0f);
  const float rstd = rsqrtf(var + EPSV);
  const int growo = rowbase + lrow;
  if (growo < n) {
    float* op = out + (long long)growo * HDIM + l16 * 8;
#pragma unroll
    for (int t = 0; t < 8; ++t)
      op[t] = (vals[t] - mu) * rstd * gam[l16 * 8 + t] + bet[l16 * 8 + t];
  }
}

// ---------------------------------------------------------------------------
extern "C" void kernel_launch(void* const* d_in, const int* in_sizes, int n_in,
                              void* d_out, int out_size, void* d_ws, size_t ws_size,
                              hipStream_t stream) {
  const float* x  = (const float*)d_in[0];   // [n, 128]
  const int*   ei = (const int*)  d_in[1];   // [2, E] (int32: JAX x64 off)
  const float* ew = (const float*)d_in[2];   // [E]
  const float* Wn = (const float*)d_in[3];   // [3, 128, 128]
  const float* Wr = (const float*)d_in[4];   // [3, 128, 128]
  const float* br = (const float*)d_in[5];   // [3, 128]
  const float* lg = (const float*)d_in[6];   // [3, 128]
  const float* lb = (const float*)d_in[7];   // [3, 128]
  float* out = (float*)d_out;

  const int n     = in_sizes[0] / HDIM;
  const int nedge = in_sizes[2];
  const int* srcp = ei;
  const int* dstp = ei + nedge;

  // workspace layout: agg[n*H] | sumw[n] | hA[n*H] | hB[n*H]
  float* agg  = (float*)d_ws;
  float* sumw = agg  + (long long)n * HDIM;
  float* hA   = sumw + n;
  float* hB   = hA   + (long long)n * HDIM;

  const long long zcnt = (long long)n * HDIM + n;   // agg+sumw contiguous
  const int zgrid = (int)((zcnt + 255) / 256);
  const int sgrid = (nedge + 7) / 8;                // 8 waves / block, 1 edge / wave
  const int ggrid = (n + 15) / 16;

  for (int l = 0; l < 3; ++l) {
    const float* hin  = (l == 0) ? x : ((l == 1) ? hA : hB);
    float*       hout = (l == 2) ? out : ((l == 0) ? hA : hB);
    gnn_zero   <<<zgrid, 256, 0, stream>>>(agg, zcnt);
    gnn_scatter<<<sgrid, 256, 0, stream>>>(hin, srcp, dstp, ew, agg, sumw, nedge);
    gnn_fused  <<<ggrid, 256, 0, stream>>>(
        hin, agg, sumw,
        Wn + (long long)l * HDIM * HDIM,
        Wr + (long long)l * HDIM * HDIM,
        br + l * HDIM, lg + l * HDIM, lb + l * HDIM,
        hout, n);
  }
}
